// VQVAE_17557826306286
// MI455X (gfx1250) — matrix operand order repaired and checked by reading, not compiled, not run on hardware
//
#include <hip/hip_runtime.h>
#include <hip/hip_bf16.h>

// ---------------------------------------------------------------------------
// MI455X (gfx1250) VQ-VAE forward.
//  * All convs / transposed-convs as implicit GEMM on v_wmma_f32_16x16x32_bf16
//    - 256 threads / 8 waves, block tile 128(M) x 64(N), K-step 32
//    - each wave: 2 A-frags x 2 B-frags -> 4 WMMA per K-step (frag reuse)
//    - conv geometry fully templated -> address math is shifts/constants
//  * Weight slab staged LDS-side by the Tensor Data Mover (TENSOR_LOAD_TO_LDS,
//    fenced with s_wait_tensorcnt) when the builtin exists; manual fallback.
//  * VQ distance matrix (131072x512, K=32) as native single-WMMA K-step GEMM
//  * bf16 NHWC activations, f32 accumulate, fused bias+ReLU on store
// ---------------------------------------------------------------------------

typedef __bf16 bf16;
typedef __bf16 v16bf __attribute__((ext_vector_type(16)));
typedef float  v8f   __attribute__((ext_vector_type(8)));
typedef unsigned int u32x4 __attribute__((ext_vector_type(4)));
typedef int          i32x4 __attribute__((ext_vector_type(4)));
typedef int          i32x8 __attribute__((ext_vector_type(8)));

#define N_EMBED 512
#define EDIM    32

// ---- TDM availability probe (compile-safe) --------------------------------
#if defined(__has_builtin)
#  if __has_builtin(__builtin_amdgcn_tensor_load_to_lds) && \
      __has_builtin(__builtin_amdgcn_s_wait_tensorcnt)
#    define USE_TDM 1
#  endif
#endif
#ifndef USE_TDM
#  define USE_TDM 0
#endif
#if defined(__has_include)
#  if __has_include(<hip/amd_detail/amd_gfx1250_TDM.h>)
#    define TDM_SIX_ARGS 1   // therock/clang-23 header lane: 6-arg builtin
#  endif
#endif
#ifndef TDM_SIX_ARGS
#  define TDM_SIX_ARGS 0
#endif

#if USE_TDM
// Issue a 2D TENSOR_LOAD_TO_LDS: tile 64(x) x 32(y) bf16 elements from a
// row-major [rows][rowStride] matrix starting at (k0, n0); columns beyond
// (Opad - n0) zero-fill via tensor OOB semantics. One issue per wave.
__device__ __forceinline__ void tdm_load_btile(const bf16* gbase, unsigned ldsAddr,
                                               int n0, int k0, int rowStride,
                                               int colsLeft, int rowsLeft) {
    unsigned long long ga = (unsigned long long)(__UINTPTR_TYPE__)gbase +
                            2ull * ((unsigned long long)(unsigned)k0 * (unsigned)rowStride +
                                    (unsigned)n0);
    // D# group0: count=1 | lds_addr | global_addr(57) | type=2
    u32x4 g0;
    g0[0] = 1u;                                   // count=1, user desc
    g0[1] = ldsAddr;                              // LDS byte address
    g0[2] = (unsigned)ga;                         // global addr [31:0]
    g0[3] = ((unsigned)(ga >> 32) & 0x01FFFFFFu)  // global addr [56:32]
            | 0x80000000u;                        // type=2 ("image")
    // D# group1: data_size=2B; tensor_dim0/1 = remaining extent from tile
    // origin (drives OOB zero-fill); tile 64x32; dim0 stride = rowStride.
    unsigned td0 = (unsigned)colsLeft;
    unsigned td1 = (unsigned)rowsLeft;
    i32x8 g1;
    g1[0] = (int)(1u << 16);                      // data_size = 1 -> 2 bytes
    g1[1] = (int)((td0 & 0xffffu) << 16);         // tensor_dim0 [15:0]
    g1[2] = (int)((td0 >> 16) | ((td1 & 0xffffu) << 16));
    g1[3] = (int)((td1 >> 16) | (64u << 16));     // tile_dim0 = 64
    g1[4] = (int)32u;                             // tile_dim1 = 32, tile_dim2 = 0
    g1[5] = (int)(unsigned)rowStride;             // tensor_dim0_stride (elements)
    g1[6] = 0;
    g1[7] = 0;
    i32x4 gz = {0, 0, 0, 0};
#if TDM_SIX_ARGS
    i32x8 gz8 = {0, 0, 0, 0, 0, 0, 0, 0};
    __builtin_amdgcn_tensor_load_to_lds(g0, g1, gz, gz, gz8, 0);
#else
    __builtin_amdgcn_tensor_load_to_lds(g0, g1, gz, gz, 0);
#endif
}
#endif // USE_TDM

// ---------------------------------------------------------------------------
// Weight repack: OIHW (or ConvTranspose IOHW + spatial flip) fp32 ->
// bf16 B-matrix [Kpad][Opad], K ordered (kh, kw, ci) with ci fastest.
// ---------------------------------------------------------------------------
__global__ void pack_weight_kernel(const float* __restrict__ w, bf16* __restrict__ dst,
                                   int O, int I, int KH, int KW,
                                   int Kpad, int Opad, int transposed) {
    long idx = (long)blockIdx.x * blockDim.x + threadIdx.x;
    long total = (long)Kpad * Opad;
    if (idx >= total) return;
    int o = (int)(idx % Opad);
    int k = (int)(idx / Opad);
    int K = KH * KW * I;
    float v = 0.f;
    if (o < O && k < K) {
        int ci = k % I;
        int t  = k / I;
        int kw = t % KW;
        int kh = t / KW;
        if (transposed) // torch ConvTranspose2d weight (in,out,kh,kw), flipped
            v = w[(((long)ci * O + o) * KH + (KH - 1 - kh)) * KW + (KW - 1 - kw)];
        else            // torch Conv2d weight (out,in,kh,kw)
            v = w[(((long)o * I + ci) * KH + kh) * KW + kw];
    }
    dst[idx] = (bf16)v;
}

// embed [512][32] fp32 -> Emat bf16 [K=32][N=512] (B-matrix layout) + ||e||^2
__global__ void pack_embed_kernel(const float* __restrict__ embed,
                                  bf16* __restrict__ Emat, float* __restrict__ enorm) {
    int n = blockIdx.x * blockDim.x + threadIdx.x;
    if (n >= N_EMBED) return;
    float s = 0.f;
#pragma unroll
    for (int k = 0; k < EDIM; ++k) {
        float v = embed[n * EDIM + k];
        s += v * v;
        Emat[k * N_EMBED + n] = (bf16)v;
    }
    enorm[n] = s;
}

__global__ void nchw_to_nhwc_bf16_kernel(const float* __restrict__ in, bf16* __restrict__ out,
                                         int B, int C, int H, int W) {
    long idx = (long)blockIdx.x * blockDim.x + threadIdx.x;
    long total = (long)B * C * H * W;
    if (idx >= total) return;
    int c = (int)(idx % C); long t = idx / C;
    int w = (int)(t % W);   t /= W;
    int h = (int)(t % H);   t /= H;
    int b = (int)t;
    out[idx] = (bf16)in[(((long)b * C + c) * H + h) * W + w];
}

__global__ void zero_loss_kernel(float* l) { *l = 0.f; }

__global__ void finalize_loss_kernel(const float* __restrict__ acc, float* __restrict__ out) {
    out[0] = 1.25f * acc[0] / 4194304.0f; // (1+0.25) * mean over 131072*32 elems
}

// ---------------------------------------------------------------------------
// Implicit-GEMM conv / convT (lhs-dilated gather) on WMMA bf16.
// Geometry is compile-time: divisions fold to shifts/magic multiplies.
//   ph = ho*STRIDE - PAD + kh ; valid iff ph>=0, ph%DIL==0, ph/DIL < H
// ---------------------------------------------------------------------------
struct ConvDims {
    const bf16*  in;    // NHWC bf16 [B,H,W,Cin]
    const bf16*  wmat;  // [KPAD][Opad] bf16
    const float* bias;  // [Cout]
    void*        out;   // bf16 NHWC  or  fp32 NCHW (final layer)
    int B, H, W;
    int Ho, Wo, Cout, Opad;
};

template <int CIN, int KH, int KW, int STRIDE, int DIL, int PAD, bool RELU, bool FINAL_F32_NCHW>
__global__ __launch_bounds__(256) void conv_wmma(ConvDims p) {
    constexpr int K    = KH * KW * CIN;
    constexpr int KPAD = (K + 31) & ~31;
    constexpr int ASTR = 34;                 // A row stride (gcd(17,64)=1 -> no bank conflicts)

    __shared__ bf16 As[128 * ASTR];          // 128 rows x 32 K
    __shared__ bf16 Bs[32 * 64];             // 32 K x 64 N (TDM-compatible layout)

    const int tid  = threadIdx.x;
    const int lane = tid & 31;
    const int wave = tid >> 5;               // 0..7
    const int wm   = wave & 3;               // 4 M-strips of 32 rows
    const int wn   = wave >> 2;              // 2 N-strips of 32 cols

    const long M  = (long)p.B * p.Ho * p.Wo;
    const long m0 = (long)blockIdx.x * 128;
    const int  n0 = blockIdx.y * 64;

    // A-loader: thread fills row rA, 16 consecutive k
    const int rA = tid >> 1;
    const int kA = (tid & 1) * 16;
    long mA = m0 + rA;
    int bA = 0, hoA = 0, woA = 0;
    const bool mAok = mA < M;
    if (mAok) {
        long t = mA;
        woA = (int)(t % p.Wo); t /= p.Wo;
        hoA = (int)(t % p.Ho); t /= p.Ho;
        bA  = (int)t;
    }

#if USE_TDM
    const unsigned bsAddr = (unsigned)(__UINTPTR_TYPE__)(void*)Bs;
#endif

    v8f acc[2][2];
#pragma unroll
    for (int mt = 0; mt < 2; ++mt)
#pragma unroll
        for (int nt = 0; nt < 2; ++nt)
            acc[mt][nt] = v8f{0.f, 0.f, 0.f, 0.f, 0.f, 0.f, 0.f, 0.f};

    for (int k0 = 0; k0 < KPAD; k0 += 32) {
        __syncthreads();
#if USE_TDM
        if (wave == 0)
            tdm_load_btile(p.wmat, bsAddr, n0, k0, p.Opad, p.Opad - n0, KPAD - k0);
#else
#pragma unroll
        for (int j = 0; j < 8; ++j) {        // manual B staging fallback
            int q  = tid * 8 + j;
            int kk = q >> 6, nn = q & 63;
            int n  = n0 + nn;
            bf16 v = (bf16)0.f;
            if (n < p.Opad) v = p.wmat[(long)(k0 + kk) * p.Opad + n];
            Bs[kk * 64 + nn] = v;
        }
#endif
        // ---- stage A (im2col gather; all index math compile-time) ----
#pragma unroll
        for (int j = 0; j < 16; ++j) {
            int k = k0 + kA + j;
            float v = 0.f;
            if (mAok && k < K) {
                int ci = k % CIN;
                int t  = k / CIN;
                int kw = t % KW;
                int kh = t / KW;
                int ph = hoA * STRIDE - PAD + kh;
                int pw = woA * STRIDE - PAD + kw;
                if (ph >= 0 && pw >= 0 && (ph % DIL) == 0 && (pw % DIL) == 0) {
                    int ih = ph / DIL, iw = pw / DIL;
                    if (ih < p.H && iw < p.W)
                        v = (float)p.in[(((long)bA * p.H + ih) * p.W + iw) * CIN + ci];
                }
            }
            As[rA * ASTR + kA + j] = (bf16)v;
        }
#if USE_TDM
        if (wave == 0)
            __builtin_amdgcn_s_wait_tensorcnt(0);
#endif
        __syncthreads();

        // ---- fragments (ISA 7.12.2 layouts) ----
        const int hl  = lane >> 4;
        const int l15 = lane & 15;
        const int kb  = hl * 8;
        v16bf aF[2], bF[2];
#pragma unroll
        for (int mt = 0; mt < 2; ++mt) {
            const int am = wm * 32 + mt * 16 + l15;
#pragma unroll
            for (int i = 0; i < 8; ++i) {
                aF[mt][i]     = As[am * ASTR + kb + i];
                aF[mt][8 + i] = As[am * ASTR + 16 + kb + i];
            }
        }
#pragma unroll
        for (int nt = 0; nt < 2; ++nt) {
            const int bn = wn * 32 + nt * 16 + l15;
#pragma unroll
            for (int i = 0; i < 16; ++i)
                bF[nt][i] = Bs[((i >> 3) * 16 + hl * 8 + (i & 7)) * 64 + bn];
        }
#pragma unroll
        for (int mt = 0; mt < 2; ++mt)
#pragma unroll
            for (int nt = 0; nt < 2; ++nt)
                acc[mt][nt] = __builtin_amdgcn_wmma_f32_16x16x32_bf16(
                    false, aF[mt], false, bF[nt], (short)0, acc[mt][nt], false, false);
    }

    // ---- epilogue: bias + relu + store ----
    const int hl  = lane >> 4;
    const int l15 = lane & 15;
#pragma unroll
    for (int nt = 0; nt < 2; ++nt) {
        const int n = n0 + wn * 32 + nt * 16 + l15;
        if (n < p.Cout) {
            const float bv = p.bias[n];
#pragma unroll
            for (int mt = 0; mt < 2; ++mt) {
#pragma unroll
                for (int r = 0; r < 8; ++r) {
                    long m = m0 + wm * 32 + mt * 16 + hl * 8 + r;
                    if (m < M) {
                        float v = acc[mt][nt][r] + bv;
                        if (RELU) v = v > 0.f ? v : 0.f;
                        if (FINAL_F32_NCHW) {
                            long t = m;
                            int wo = (int)(t % p.Wo); t /= p.Wo;
                            int ho = (int)(t % p.Ho); t /= p.Ho;
                            int bb = (int)t;
                            ((float*)p.out)[(((long)bb * p.Cout + n) * p.Ho + ho) * p.Wo + wo] = v;
                        } else {
                            ((bf16*)p.out)[m * p.Cout + n] = (bf16)v;
                        }
                    }
                }
            }
        }
    }
}

// ---------------------------------------------------------------------------
// Vector quantizer: one wave per 16 z-rows; dot via WMMA (K=32 native),
// per-row argmin, shfl_xor reduce, loss accumulation, z_q gather.
// ---------------------------------------------------------------------------
__global__ __launch_bounds__(32) void vq_kernel(const bf16* __restrict__ z,
                                                const bf16* __restrict__ Emat,
                                                const float* __restrict__ enorm,
                                                const float* __restrict__ embed,
                                                bf16* __restrict__ zq,
                                                float* __restrict__ lossAccum,
                                                long Nrows) {
    __shared__ float znS[16];
    __shared__ int   idxS[16];

    const int lane = threadIdx.x;
    const int hl   = lane >> 4;
    const int l15  = lane & 15;
    const int kb   = hl * 8;
    const long row0 = (long)blockIdx.x * 16;

    v16bf a;
    float part = 0.f;
#pragma unroll
    for (int i = 0; i < 8; ++i) {
        float x0 = (float)z[(row0 + l15) * EDIM + kb + i];
        float x1 = (float)z[(row0 + l15) * EDIM + 16 + kb + i];
        a[i]     = (bf16)x0;
        a[8 + i] = (bf16)x1;
        part += x0 * x0 + x1 * x1;
    }
    part += __shfl_xor(part, 16, 32);
    if (lane < 16) znS[lane] = part;
    __syncthreads();

    float best[8];
    int   bidx[8];
#pragma unroll
    for (int r = 0; r < 8; ++r) { best[r] = 3.4e38f; bidx[r] = 0; }

    for (int n0 = 0; n0 < N_EMBED; n0 += 16) {
        v16bf b;
#pragma unroll
        for (int i = 0; i < 16; ++i)
            b[i] = Emat[((i >> 3) * 16 + hl * 8 + (i & 7)) * N_EMBED + n0 + l15];
        v8f dot = {0.f, 0.f, 0.f, 0.f, 0.f, 0.f, 0.f, 0.f};
        dot = __builtin_amdgcn_wmma_f32_16x16x32_bf16(false, a, false, b,
                                                      (short)0, dot, false, false);
        const float en = enorm[n0 + l15];
#pragma unroll
        for (int r = 0; r < 8; ++r) {
            float d = znS[hl * 8 + r] + en - 2.f * dot[r];
            if (d < best[r]) { best[r] = d; bidx[r] = n0 + l15; }
        }
    }

#pragma unroll
    for (int off = 8; off >= 1; off >>= 1) {
#pragma unroll
        for (int r = 0; r < 8; ++r) {
            float ob = __shfl_xor(best[r], off, 32);
            int   oi = __shfl_xor(bidx[r], off, 32);
            if (ob < best[r] || (ob == best[r] && oi < bidx[r])) { best[r] = ob; bidx[r] = oi; }
        }
    }
    if (l15 == 0) {
        float ls = 0.f;
#pragma unroll
        for (int r = 0; r < 8; ++r) {
            idxS[hl * 8 + r] = bidx[r];
            ls += best[r] > 0.f ? best[r] : 0.f;
        }
        atomicAdd(lossAccum, ls);
    }
    __syncthreads();

#pragma unroll
    for (int t = 0; t < 16; ++t) {
        long row = row0 + t;
        if (row < Nrows)
            zq[row * EDIM + lane] = (bf16)embed[(long)idxS[t] * EDIM + lane];
    }
}

// ---------------------------------------------------------------------------
// Host-side orchestration
// ---------------------------------------------------------------------------
extern "C" void kernel_launch(void* const* d_in, const int* in_sizes, int n_in,
                              void* d_out, int out_size, void* d_ws, size_t ws_size,
                              hipStream_t stream) {
    (void)in_sizes; (void)n_in; (void)out_size; (void)ws_size;

    const float* x     = (const float*)d_in[0];
    const float* e_w1  = (const float*)d_in[1];
    const float* e_b1  = (const float*)d_in[2];
    const float* e_w2  = (const float*)d_in[3];
    const float* e_b2  = (const float*)d_in[4];
    const float* e_w3  = (const float*)d_in[5];
    const float* e_b3  = (const float*)d_in[6];
    const float* embed = (const float*)d_in[7];
    const float* d_w1  = (const float*)d_in[8];
    const float* d_b1  = (const float*)d_in[9];
    const float* d_w2  = (const float*)d_in[10];
    const float* d_b2  = (const float*)d_in[11];
    const float* d_w3  = (const float*)d_in[12];
    const float* d_b3  = (const float*)d_in[13];
    float* out = (float*)d_out;

    char* ws = (char*)d_ws;
    size_t off = 0;
    auto alloc = [&](size_t bytes) {
        size_t o = off;
        off = (off + bytes + 255) & ~(size_t)255;
        return o;
    };
    const size_t o_wc1 = alloc((size_t)64   * 64  * 2);
    const size_t o_wc2 = alloc((size_t)1024 * 128 * 2);
    const size_t o_wc3 = alloc((size_t)1152 * 32  * 2);
    const size_t o_wd1 = alloc((size_t)288  * 128 * 2);
    const size_t o_wd2 = alloc((size_t)2048 * 64  * 2);
    const size_t o_wd3 = alloc((size_t)1024 * 16  * 2);
    const size_t o_emt = alloc((size_t)EDIM * N_EMBED * 2);
    const size_t o_enm = alloc((size_t)N_EMBED * 4);
    const size_t o_lss = alloc(256);
    const size_t o_H1  = alloc((size_t)32 * 128 * 128 * 64 * 2);  // h1, reused as d2-out
    const size_t o_H2  = alloc((size_t)32 * 64  * 64  * 128 * 2); // h2, reused as d1-out
    const size_t o_X   = alloc((size_t)32 * 256 * 256 * 3   * 2); // x bf16, reused as z_q
    const size_t o_Z   = alloc((size_t)32 * 64  * 64  * 32  * 2); // z

    bf16* wc1 = (bf16*)(ws + o_wc1);  bf16* wc2 = (bf16*)(ws + o_wc2);
    bf16* wc3 = (bf16*)(ws + o_wc3);  bf16* wd1 = (bf16*)(ws + o_wd1);
    bf16* wd2 = (bf16*)(ws + o_wd2);  bf16* wd3 = (bf16*)(ws + o_wd3);
    bf16* Emat = (bf16*)(ws + o_emt);
    float* enorm = (float*)(ws + o_enm);
    float* lossA = (float*)(ws + o_lss);
    bf16* bufH1 = (bf16*)(ws + o_H1);
    bf16* bufH2 = (bf16*)(ws + o_H2);
    bf16* bufX  = (bf16*)(ws + o_X);
    bf16* bufZ  = (bf16*)(ws + o_Z);

    auto gpk = [](long total) { return dim3((unsigned)((total + 255) / 256)); };

    // ---- packing ----
    pack_weight_kernel<<<gpk(64L * 64), 256, 0, stream>>>(e_w1, wc1, 64, 3, 4, 4, 64, 64, 0);
    pack_weight_kernel<<<gpk(1024L * 128), 256, 0, stream>>>(e_w2, wc2, 128, 64, 4, 4, 1024, 128, 0);
    pack_weight_kernel<<<gpk(1152L * 32), 256, 0, stream>>>(e_w3, wc3, 32, 128, 3, 3, 1152, 32, 0);
    pack_weight_kernel<<<gpk(288L * 128), 256, 0, stream>>>(d_w1, wd1, 128, 32, 3, 3, 288, 128, 1);
    pack_weight_kernel<<<gpk(2048L * 64), 256, 0, stream>>>(d_w2, wd2, 64, 128, 4, 4, 2048, 64, 1);
    pack_weight_kernel<<<gpk(1024L * 16), 256, 0, stream>>>(d_w3, wd3, 3, 64, 4, 4, 1024, 16, 1);
    pack_embed_kernel<<<2, 256, 0, stream>>>(embed, Emat, enorm);
    zero_loss_kernel<<<1, 1, 0, stream>>>(lossA);
    nchw_to_nhwc_bf16_kernel<<<gpk(32L * 3 * 256 * 256), 256, 0, stream>>>(x, bufX, 32, 3, 256, 256);

    // ---- encoder ----
    {   // conv1: [32,256,256,3] -> [32,128,128,64], k4 s2 p1, relu  (M=524288)
        ConvDims q{bufX, wc1, e_b1, (void*)bufH1, 32, 256, 256, 128, 128, 64, 64};
        conv_wmma<3, 4, 4, 2, 1, 1, true, false><<<dim3(4096, 1), 256, 0, stream>>>(q);
    }
    {   // conv2: -> [32,64,64,128], k4 s2 p1, relu  (M=131072)
        ConvDims q{bufH1, wc2, e_b2, (void*)bufH2, 32, 128, 128, 64, 64, 128, 128};
        conv_wmma<64, 4, 4, 2, 1, 1, true, false><<<dim3(1024, 2), 256, 0, stream>>>(q);
    }
    {   // conv3: -> z [32,64,64,32], k3 s1 p1, relu
        ConvDims q{bufH2, wc3, e_b3, (void*)bufZ, 32, 64, 64, 64, 64, 32, 32};
        conv_wmma<128, 3, 3, 1, 1, 1, true, false><<<dim3(1024, 1), 256, 0, stream>>>(q);
    }

    // ---- vector quantizer (z_q into bufX region, loss accum) ----
    vq_kernel<<<8192, 32, 0, stream>>>(bufZ, Emat, enorm, embed, bufX, lossA, 131072L);

    // ---- decoder ----
    {   // convT1 k3 s1 p1 == conv(flip) pad 1: z_q -> [32,64,64,128], relu
        ConvDims q{bufX, wd1, d_b1, (void*)bufH2, 32, 64, 64, 64, 64, 128, 128};
        conv_wmma<32, 3, 3, 1, 1, 1, true, false><<<dim3(1024, 2), 256, 0, stream>>>(q);
    }
    {   // convT2 k4 s2 p1 == conv(flip), lhs-dil 2, pad 2 -> [32,128,128,64], relu
        ConvDims q{bufH2, wd2, d_b2, (void*)bufH1, 32, 64, 64, 128, 128, 64, 64};
        conv_wmma<128, 4, 4, 1, 2, 2, true, false><<<dim3(4096, 1), 256, 0, stream>>>(q);
    }
    {   // convT3 k4 s2 p1 -> x_recon [32,3,256,256] fp32 NCHW, no relu (M=2097152)
        ConvDims q{bufH1, wd3, d_b3, (void*)out, 32, 128, 128, 256, 256, 3, 16};
        conv_wmma<64, 4, 4, 1, 2, 2, false, true><<<dim3(16384, 1), 256, 0, stream>>>(q);
    }

    finalize_loss_kernel<<<1, 1, 0, stream>>>(lossA, out + 6291456);
}
